// MambaBlock_75479755260591
// MI455X (gfx1250) — matrix operand order repaired and checked by reading, not compiled, not run on hardware
//
#include <hip/hip_runtime.h>
#include <hip/hip_bf16.h>
#include <math.h>

typedef __attribute__((ext_vector_type(16))) __bf16 v16bf;
typedef __attribute__((ext_vector_type(8)))  __bf16 v8bf;
typedef __attribute__((ext_vector_type(8)))  float  v8f;

static constexpr int   Bb    = 8;
static constexpr int   Ls    = 2048;
static constexpr int   Dd    = 1024;
static constexpr int   Mrows = Bb * Ls;   // 16384
static constexpr float LN_EPS = 1e-5f;

// ---------------------------------------------------------------------------
// f32 -> bf16 conversion (weights, done once per launch)
// ---------------------------------------------------------------------------
__global__ void cvt_f32_bf16(const float* __restrict__ src,
                             __bf16* __restrict__ dst, int n) {
    int i = blockIdx.x * blockDim.x + threadIdx.x;
    if (i < n) dst[i] = (__bf16)src[i];
}

// ---------------------------------------------------------------------------
// LayerNorm over D=1024, one 256-thread block per row; writes f32 + bf16
// ---------------------------------------------------------------------------
__global__ void __launch_bounds__(256)
layernorm_kernel(const float* __restrict__ x, const float* __restrict__ gamma,
                 const float* __restrict__ beta, float* __restrict__ xn,
                 __bf16* __restrict__ xn_bf) {
    const int row  = blockIdx.x;
    const int tid  = threadIdx.x;
    const int lane = tid & 31;
    const int wave = tid >> 5;
    const float* xr = x + (size_t)row * Dd;

    __shared__ float sh[8];

    float v[4];
    float s = 0.f;
    #pragma unroll
    for (int i = 0; i < 4; ++i) { v[i] = xr[tid + i * 256]; s += v[i]; }
    #pragma unroll
    for (int off = 16; off > 0; off >>= 1) s += __shfl_down(s, off, 32);
    if (lane == 0) sh[wave] = s;
    __syncthreads();
    if (tid == 0) {
        float t = 0.f;
        #pragma unroll
        for (int i = 0; i < 8; ++i) t += sh[i];
        sh[0] = t * (1.0f / Dd);
    }
    __syncthreads();
    const float mu = sh[0];

    float vs = 0.f;
    #pragma unroll
    for (int i = 0; i < 4; ++i) { float d = v[i] - mu; vs += d * d; }
    #pragma unroll
    for (int off = 16; off > 0; off >>= 1) vs += __shfl_down(vs, off, 32);
    __syncthreads();                 // everyone has read sh[0] (mu)
    if (lane == 0) sh[wave] = vs;
    __syncthreads();
    if (tid == 0) {
        float t = 0.f;
        #pragma unroll
        for (int i = 0; i < 8; ++i) t += sh[i];
        sh[0] = rsqrtf(t * (1.0f / Dd) + LN_EPS);
    }
    __syncthreads();
    const float rstd = sh[0];

    #pragma unroll
    for (int i = 0; i < 4; ++i) {
        int d = tid + i * 256;
        float y = (v[i] - mu) * rstd * gamma[d] + beta[d];
        xn[(size_t)row * Dd + d]    = y;
        xn_bf[(size_t)row * Dd + d] = (__bf16)y;
    }
}

// ---------------------------------------------------------------------------
// WMMA bf16 GEMM:  C[m,n] = sum_k A[m,k] * W[n,k]  (+ bias[n]) (+ addend[m,n])
// Per-wave tile: 32x64 (2 M-fragments x 4 N-fragments, 8 f32x8 accumulators)
//  -> 8 v_wmma per K-step against 12 b128 loads (~21 FLOP/B operand intensity).
// Fragment layouts per CDNA5 ISA §7.12.2:
//   A (16x32 bf16): lane = M-row (lane&15); lanes 0-15 hold K 0-7 / 16-23,
//                   lanes 16-31 hold K 8-15 / 24-31  -> two 16B loads.
//   B (32x16 bf16): lane&15 = N-col; lanes 0-15 hold K 0-7 / 16-23,
//                   lanes 16-31 hold K 8-15 / 24-31  -> two 16B loads.
//   C (16x16 f32):  N = lane&15; VGPR r holds M = r + (lane>>4)*8.
// ---------------------------------------------------------------------------
__device__ __forceinline__ v16bf load_a_frag(const __bf16* __restrict__ A,
                                             int lda, size_t row, int kb, int lane) {
    const int half = lane >> 4;
    const __bf16* p = A + row * (size_t)lda + kb + half * 8;
    v8bf lo = *(const v8bf*)p;
    v8bf hi = *(const v8bf*)(p + 16);
    v16bf r;
    #pragma unroll
    for (int i = 0; i < 8; ++i) { r[i] = lo[i]; r[8 + i] = hi[i]; }
    return r;
}

__device__ __forceinline__ v16bf load_b_frag(const __bf16* __restrict__ W,
                                             int ldw, size_t col, int kb, int lane) {
    const int half = lane >> 4;
    const __bf16* p = W + col * (size_t)ldw + kb + half * 8;
    v8bf lo = *(const v8bf*)p;
    v8bf hi = *(const v8bf*)(p + 16);
    v16bf r;
    #pragma unroll
    for (int i = 0; i < 8; ++i) { r[i] = lo[i]; r[8 + i] = hi[i]; }
    return r;
}

__global__ void __launch_bounds__(256)
gemm_bf16_wmma(const __bf16* __restrict__ A, const __bf16* __restrict__ W,
               const float* __restrict__ bias, const float* __restrict__ addend,
               float* __restrict__ C, int M, int N, int K) {
    const int lane = threadIdx.x & 31;
    const int wave = threadIdx.x >> 5;
    const int m0 = blockIdx.y * 256 + wave * 32;   // 8 waves cover 256 rows
    const int n0 = blockIdx.x * 64;                // 4 x 16 columns per wave
    if (m0 >= M) return;

    const size_t arow0 = (size_t)(m0 + (lane & 15));
    const size_t arow1 = arow0 + 16;
    const v8f z = {0.f, 0.f, 0.f, 0.f, 0.f, 0.f, 0.f, 0.f};
    v8f acc[2][4];
    #pragma unroll
    for (int i = 0; i < 2; ++i)
        #pragma unroll
        for (int j = 0; j < 4; ++j) acc[i][j] = z;

    for (int kb = 0; kb < K; kb += 32) {
        // prefetch next K-tile of A into cache (global_prefetch_b8)
        if (kb + 32 < K) {
            __builtin_prefetch(A + arow0 * (size_t)K + kb + 32, 0, 0);
            __builtin_prefetch(A + arow1 * (size_t)K + kb + 32, 0, 0);
        }
        v16bf a0 = load_a_frag(A, K, arow0, kb, lane);
        v16bf a1 = load_a_frag(A, K, arow1, kb, lane);
        #pragma unroll
        for (int j = 0; j < 4; ++j) {
            v16bf b = load_b_frag(W, K, (size_t)(n0 + j * 16 + (lane & 15)), kb, lane);
            acc[0][j] = __builtin_amdgcn_wmma_f32_16x16x32_bf16(
                false, a0, false, b, (short)0, acc[0][j], false, false);
            acc[1][j] = __builtin_amdgcn_wmma_f32_16x16x32_bf16(
                false, a1, false, b, (short)0, acc[1][j], false, false);
        }
    }

    const int nlane = lane & 15;
    const int mbase = (lane >> 4) * 8;
    #pragma unroll
    for (int j = 0; j < 4; ++j) {
        const int n = n0 + j * 16 + nlane;
        const float bv = bias[n];
        #pragma unroll
        for (int i = 0; i < 2; ++i) {
            #pragma unroll
            for (int r = 0; r < 8; ++r) {
                const int m = m0 + i * 16 + mbase + r;
                const size_t idx = (size_t)m * N + n;
                float val = acc[i][j][r] + bv;
                if (addend) val += addend[idx];
                C[idx] = val;
            }
        }
    }
}

// ---------------------------------------------------------------------------
// gate/u elementwise, IN PLACE inside xp:
//   xp[row, 0:D]   : raw delta -> gate = sum_s exp(softplus(delta) * A[d,s])
//   xp[row, D:2D]  : Bg        -> u    = xn * Bg
// ---------------------------------------------------------------------------
__global__ void __launch_bounds__(256)
gate_u_kernel(float* __restrict__ xp, const float* __restrict__ xn,
              const float* __restrict__ Amat) {
    const size_t i = (size_t)blockIdx.x * blockDim.x + threadIdx.x;  // over M*D
    const size_t row = i >> 10;
    const int d = (int)(i & 1023);
    const size_t base = row * (size_t)(2 * Dd);

    const float draw = xp[base + d];
    const float bg   = xp[base + Dd + d];
    const float delta = (draw > 20.f) ? draw : log1pf(__expf(draw));

    float gate = 0.f;
    const float* Ad = Amat + (size_t)d * 16;
    #pragma unroll
    for (int s = 0; s < 16; ++s) gate += __expf(delta * Ad[s]);

    const float u = xn[i] * bg;
    xp[base + d]      = gate;
    xp[base + Dd + d] = u;
}

// ---------------------------------------------------------------------------
// Sequential cumsum over L per (b,d) chain, fused epilogue:
//   out0 = cumsum(u)*gate + xn*D_param  -> bf16 for GEMM2.
// Grid: (D/256, B); each thread owns one d-column, loads coalesced.
// ---------------------------------------------------------------------------
__global__ void __launch_bounds__(256)
scan_epilogue_kernel(const float* __restrict__ xp, const float* __restrict__ xn,
                     const float* __restrict__ Dparam, __bf16* __restrict__ out_bf) {
    const int d = blockIdx.x * 256 + threadIdx.x;
    const int b = blockIdx.y;
    const float dp = Dparam[d];
    float carry = 0.f;
    for (int l = 0; l < Ls; ++l) {
        const size_t row  = (size_t)b * Ls + l;
        const size_t base = row * (size_t)(2 * Dd);
        carry += xp[base + Dd + d];                              // u
        const float o = carry * xp[base + d]                     // h * gate
                      + xn[row * (size_t)Dd + d] * dp;           // + xn * D
        out_bf[row * (size_t)Dd + d] = (__bf16)o;
    }
}

// ---------------------------------------------------------------------------
// Launch
// ---------------------------------------------------------------------------
extern "C" void kernel_launch(void* const* d_in, const int* in_sizes, int n_in,
                              void* d_out, int out_size, void* d_ws, size_t ws_size,
                              hipStream_t stream) {
    const float* x      = (const float*)d_in[0];
    const float* w_xp   = (const float*)d_in[1];
    const float* b_xp   = (const float*)d_in[2];
    const float* Amat   = (const float*)d_in[3];
    const float* Dparam = (const float*)d_in[4];
    const float* w_out  = (const float*)d_in[5];
    const float* b_out  = (const float*)d_in[6];
    const float* gamma  = (const float*)d_in[7];
    const float* beta   = (const float*)d_in[8];

    char* ws = (char*)d_ws;
    // workspace layout (bytes):
    //   [0,              64 MiB)  xn  f32      (M x D)
    //   [64 MiB,         96 MiB)  xn_bf bf16   (M x D)  -- reused as out0_bf
    //   [96 MiB,        224 MiB)  xp  f32      (M x 2D) -- becomes gate|u in place
    //   [224 MiB, +4 MiB)         w_xproj bf16 (2D x D)
    //   [+,       +2 MiB)         w_out   bf16 (D x D)
    float*  xn    = (float*) (ws + 0);
    __bf16* xn_bf = (__bf16*)(ws + 67108864ull);
    float*  xp    = (float*) (ws + 100663296ull);
    __bf16* wx_bf = (__bf16*)(ws + 234881024ull);
    __bf16* wo_bf = (__bf16*)(ws + 239075328ull);

    cvt_f32_bf16<<<(2 * Dd * Dd + 255) / 256, 256, 0, stream>>>(w_xp, wx_bf, 2 * Dd * Dd);
    cvt_f32_bf16<<<(Dd * Dd + 255) / 256, 256, 0, stream>>>(w_out, wo_bf, Dd * Dd);

    layernorm_kernel<<<Mrows, 256, 0, stream>>>(x, gamma, beta, xn, xn_bf);

    // GEMM1: xp = xn_bf (16384x1024) * w_xprojT -> (16384x2048), + b_xproj
    gemm_bf16_wmma<<<dim3((2 * Dd) / 64, Mrows / 256), 256, 0, stream>>>(
        xn_bf, wx_bf, b_xp, nullptr, xp, Mrows, 2 * Dd, Dd);

    gate_u_kernel<<<(unsigned)((size_t)Mrows * Dd / 256), 256, 0, stream>>>(xp, xn, Amat);

    scan_epilogue_kernel<<<dim3(Dd / 256, Bb), 256, 0, stream>>>(xp, xn, Dparam, xn_bf);

    // GEMM2: out = out0_bf (16384x1024) * w_outT -> (16384x1024), + b_out + residual x
    gemm_bf16_wmma<<<dim3(Dd / 64, Mrows / 256), 256, 0, stream>>>(
        xn_bf, wo_bf, b_out, x, (float*)d_out, Mrows, Dd, Dd);
}
